// MultiHeadAttentionQuantum_65481071399444
// MI455X (gfx1250) — compile-verified, matrix operand order checked by reading, not run on hardware
//
#include <hip/hip_runtime.h>

// MI455X / gfx1250, wave32. Fused quantum-head + attention + WMMA projection.
//
// Math note: the circuit RX(x_i+theta_i) on each wire followed by the CNOT
// chain (0,1)..(6,7),(7,0) has closed-form PauliZ expectations:
//   c_i = cos(x_i + theta_i)
//   z[w] = prod_{i=0..w} c_i   (w = 1..7)
//   z[0] = prod_{i=1..7} c_i
// (Heisenberg propagation of Z through the CNOT chain.)

typedef __attribute__((ext_vector_type(16))) _Float16 v16h;
typedef __attribute__((ext_vector_type(8)))  float    v8f;

#define THREADS          128   // 4 waves (wave32)
#define TOKENS_PER_BLOCK 64    // 16 tokens per wave (one WMMA M-tile)

__global__ __launch_bounds__(THREADS)
void qattn_fused_kernel(const float* __restrict__ x,      // [T,64]
                        const float* __restrict__ theta,  // [8]
                        const float* __restrict__ W,      // [64,64] row-major [e][k]
                        const float* __restrict__ bias,   // [64]
                        float* __restrict__ out)          // [T,64]
{
    // Padded strides (72 halves / 72 floats) keep the per-lane WMMA-fragment
    // gathers spread across LDS banks (gcd(36,64)=4 -> 16 distinct banks).
    __shared__ float    zsh[16][72];    // z rows for 16 in-flight tokens
    __shared__ _Float16 Ash[64 * 72];   // attention output, f16, A operand
    __shared__ _Float16 Whs[64 * 72];   // W converted to f16, B operand [e][k]

    const int tid = threadIdx.x;

    // ---- cooperative load + f32->f16 convert of W into LDS ----
    #pragma unroll
    for (int i = 0; i < 32; ++i) {
        const int idx = tid + i * THREADS;          // 0..4095
        const int e = idx >> 6, k = idx & 63;
        Whs[e * 72 + k] = (_Float16)W[idx];
    }

    float th[8];
    #pragma unroll
    for (int i = 0; i < 8; ++i) th[i] = theta[i];   // uniform -> s_loads

    const int lt = tid >> 3;          // local token slot 0..15
    const int h  = tid & 7;           // head 0..7
    const float scale = 0.3535533905932738f;        // 1/sqrt(8)

    // =========== Phase A: quantum heads + 8x8 attention per token ===========
    for (int it = 0; it < 4; ++it) {
        const int  j     = it * 16 + lt;                       // token-in-block 0..63
        const long token = (long)blockIdx.x * TOKENS_PER_BLOCK + j;
        const float* xp  = x + token * 64 + h * 8;

        const float4 xa = *(const float4*)(xp);
        const float4 xb = *(const float4*)(xp + 4);

        float c[8];
        c[0] = __cosf(xa.x + th[0]);
        c[1] = __cosf(xa.y + th[1]);
        c[2] = __cosf(xa.z + th[2]);
        c[3] = __cosf(xa.w + th[3]);
        c[4] = __cosf(xb.x + th[4]);
        c[5] = __cosf(xb.y + th[5]);
        c[6] = __cosf(xb.z + th[6]);
        c[7] = __cosf(xb.w + th[7]);

        float z[8];
        z[1] = c[0] * c[1];
        #pragma unroll
        for (int w = 2; w < 8; ++w) z[w] = z[w - 1] * c[w];
        {   // z[0] = prod c_1..c_7
            float t = c[1] * c[2];
            t *= c[3]; t *= c[4]; t *= c[5]; t *= c[6]; t *= c[7];
            z[0] = t;
        }

        #pragma unroll
        for (int d = 0; d < 8; ++d) zsh[lt][h * 8 + d] = z[d];
        __syncthreads();

        float zr[8][8];   // all 8 head-rows of this token (static indices only)
        #pragma unroll
        for (int t = 0; t < 8; ++t)
            #pragma unroll
            for (int d = 0; d < 8; ++d)
                zr[t][d] = zsh[lt][t * 8 + d];

        // scores: own row (z[] in regs) dot every row; softmax over t
        float s[8];
        #pragma unroll
        for (int t = 0; t < 8; ++t) {
            float acc = 0.f;
            #pragma unroll
            for (int d = 0; d < 8; ++d) acc += z[d] * zr[t][d];
            s[t] = acc * scale;
        }
        float mx = s[0];
        #pragma unroll
        for (int t = 1; t < 8; ++t) mx = fmaxf(mx, s[t]);
        float e[8], sum = 0.f;
        #pragma unroll
        for (int t = 0; t < 8; ++t) { e[t] = __expf(s[t] - mx); sum += e[t]; }
        const float inv = 1.0f / sum;

        float o[8] = {0.f, 0.f, 0.f, 0.f, 0.f, 0.f, 0.f, 0.f};
        #pragma unroll
        for (int t = 0; t < 8; ++t)
            #pragma unroll
            for (int d = 0; d < 8; ++d)
                o[d] += e[t] * zr[t][d];

        #pragma unroll
        for (int d = 0; d < 8; ++d)
            Ash[j * 72 + h * 8 + d] = (_Float16)(o[d] * inv);

        __syncthreads();   // zsh reused next iter; last one publishes Ash/Whs
    }

    // =========== Phase B: [64tok,64] x W^T [64,64] + b via WMMA f16 ===========
    const int lane = tid & 31;
    const int wave = tid >> 5;                       // 0..3, one 16-row M-tile each
    const int mr   = lane & 15;
    const int kb   = (lane < 16) ? 0 : 8;            // ISA 16-bit A 16x32 layout
    const long m0  = (long)blockIdx.x * TOKENS_PER_BLOCK + wave * 16;

    // A fragments: two K-tiles of 32 (K = 64 total)
    v16h a0, a1;
    #pragma unroll
    for (int r = 0; r < 8; ++r) {
        const int k = kb + ((r < 4) ? (2 * r) : (16 + 2 * (r - 4)));
        const _Float16* ap = &Ash[(wave * 16 + mr) * 72 + k];
        a0[2 * r]     = ap[0];
        a0[2 * r + 1] = ap[1];
        a1[2 * r]     = ap[32];
        a1[2 * r + 1] = ap[33];
    }

    #pragma unroll
    for (int nt = 0; nt < 4; ++nt) {
        const int n = mr + nt * 16;                  // output feature (column)
        v16h b0, b1;                                 // B[k][n] = W[n][k]
        #pragma unroll
        for (int r = 0; r < 8; ++r) {
            const int k = kb + ((r < 4) ? (2 * r) : (16 + 2 * (r - 4)));
            const _Float16* bp = &Whs[n * 72 + k];
            b0[2 * r]     = bp[0];
            b0[2 * r + 1] = bp[1];
            b1[2 * r]     = bp[32];
            b1[2 * r + 1] = bp[33];
        }
        v8f acc = {};
        acc = __builtin_amdgcn_wmma_f32_16x16x32_f16(false, a0, false, b0,
                                                     (short)0, acc, false, false);
        acc = __builtin_amdgcn_wmma_f32_16x16x32_f16(false, a1, false, b1,
                                                     (short)0, acc, false, false);
        const float bv = bias[n];
        #pragma unroll
        for (int r = 0; r < 8; ++r) {
            const int mrow = (lane < 16) ? r : (r + 8);   // ISA C/D layout
            out[(m0 + mrow) * 64 + n] = acc[r] + bv;
        }
    }
}

extern "C" void kernel_launch(void* const* d_in, const int* in_sizes, int n_in,
                              void* d_out, int out_size, void* d_ws, size_t ws_size,
                              hipStream_t stream) {
    const float* x     = (const float*)d_in[0];   // [32,512,64]
    const float* theta = (const float*)d_in[1];   // [8]
    const float* W     = (const float*)d_in[2];   // [64,64]
    const float* b     = (const float*)d_in[3];   // [64]
    float* out         = (float*)d_out;           // [32,512,64]

    const int T    = in_sizes[0] / 64;            // 16384 tokens
    const int grid = T / TOKENS_PER_BLOCK;        // 256 blocks

    qattn_fused_kernel<<<grid, THREADS, 0, stream>>>(x, theta, W, b, out);
}
